// GCN_58308476010759
// MI455X (gfx1250) — matrix operand order repaired and checked by reading, not compiled, not run on hardware
//
#include <hip/hip_runtime.h>
#include <math.h>

// ---------------------------------------------------------------------------
// GCN + top-k pooling pipeline for MI455X (gfx1250, wave32, WMMA).
// Aggregation GEMMs: f16-in / f32-acc v_wmma_f32_16x16x32_f16. A and Z are
// zero-padded to K-multiples of 32 so the WMMA loop is branch-free b128 loads.
// Z is stored transposed (F x N32) so B-fragments are contiguous per lane.
// ---------------------------------------------------------------------------

typedef __attribute__((ext_vector_type(16))) _Float16 v16h;
typedef __attribute__((ext_vector_type(8)))  _Float16 v8h;
typedef __attribute__((ext_vector_type(8)))  float    v8f;

#define BATCH 256
#define NEG_INF (-3.0e38f)

// ---------------- 1. Build dense symmetric adjacency from packed triu -------
__global__ void build_adj(const float* __restrict__ data, float* __restrict__ A,
                          int N, int P, size_t total) {
  size_t t = (size_t)blockIdx.x * blockDim.x + threadIdx.x;
  if (t >= total) return;
  int j = (int)(t % N);
  size_t r = t / N;
  int i = (int)(r % N);
  int b = (int)(r / N);
  float v = 0.0f;
  if (i != j) {
    int a = i < j ? i : j;
    int c = i < j ? j : i;
    int pidx = a * (2 * N - a - 1) / 2 + (c - a - 1);   // triu_indices(N,1) order
    v = data[(size_t)b * P + pidx];
  }
  A[t] = v;
}

// ---------------- 2. deg = 1 + sum_j A[i,j];  dis = rsqrt(deg) --------------
__global__ void deg_dis(const float* __restrict__ A, float* __restrict__ dis,
                        int N, size_t total) {
  size_t t = (size_t)blockIdx.x * blockDim.x + threadIdx.x;
  if (t >= total) return;                 // t = b*N + i
  const float* row = A + t * (size_t)N;
  float s = 1.0f;                          // self loop
  for (int j = 0; j < N; ++j) s += row[j];
  dis[t] = (s > 0.0f) ? rsqrtf(s) : 0.0f;
}

// ---------------- 3. Ah = f16(A + I), zero-padded to (N32 x N32) -----------
__global__ void a_to_half_pad(const float* __restrict__ A, _Float16* __restrict__ Ah,
                              int N, int N32, size_t total) {
  size_t t = (size_t)blockIdx.x * blockDim.x + threadIdx.x;
  if (t >= total) return;                 // t = (b*N32 + i)*N32 + j
  int j = (int)(t % N32);
  size_t r = t / N32;
  int i = (int)(r % N32);
  int b = (int)(r / N32);
  float v = 0.0f;
  if (i < N && j < N)
    v = A[((size_t)b * N + i) * N + j] + ((i == j) ? 1.0f : 0.0f);
  Ah[t] = (_Float16)v;
}

// ---------------- 4. Zt[b,f,k] = dis_k * (X @ W)[k,f]  (f16, K padded) -----
__global__ void xw_scale_t(const float* __restrict__ X, const float* __restrict__ W,
                           const float* __restrict__ dis, _Float16* __restrict__ Zt,
                           int N, int N32, int Fin, int Fout, size_t total) {
  size_t t = (size_t)blockIdx.x * blockDim.x + threadIdx.x;
  if (t >= total) return;                 // t = (b*Fout + f)*N32 + k
  int k = (int)(t % N32);
  size_t r = t / N32;
  int f = (int)(r % Fout);
  int b = (int)(r / Fout);
  float val = 0.0f;
  if (k < N) {
    float acc = 0.0f;
    if (X) {
      const float* xr = X + ((size_t)b * N + k) * Fin;
      for (int c = 0; c < Fin; ++c) acc += xr[c] * W[c * Fout + f];
    } else {
      acc = W[f];                          // layer 1: X == ones, Fin == 1
    }
    val = acc * dis[(size_t)b * N + k];
  }
  Zt[t] = (_Float16)val;
}

// ---------------- 5. H = relu(dis_i * (Ah @ Z) + bias)  [WMMA] -------------
// One wave per 16-row strip; A-fragment reused across NT = F/16 feature tiles.
// All fragment loads are contiguous 16B (b128) — layouts per ISA 7.12.2.
template <int NT>
__global__ void __launch_bounds__(32)
gcn_agg_wmma(const _Float16* __restrict__ Ah,   // (B, N32, N32) padded A+I
             const _Float16* __restrict__ Zt,   // (B, NT*16, N32) transposed
             const float* __restrict__ dis,     // (B, N)
             const float* __restrict__ bias,    // (NT*16)
             float* __restrict__ H,             // (B, N, NT*16)
             int N, int N32) {
  constexpr int F = NT * 16;
  const int b    = blockIdx.y;
  const int tm   = blockIdx.x;               // node-row tile
  const int lane = threadIdx.x;              // 0..31
  const int mloc = lane & 15;
  const int half = lane >> 4;

  const _Float16* Arow = Ah + ((size_t)b * N32 + (tm * 16 + mloc)) * N32;
  const _Float16* Zcol = Zt + ((size_t)b * F + mloc) * N32;

  union Frag { v16h v; v8h h8[2]; };
  v8f acc[NT];
  #pragma unroll
  for (int t = 0; t < NT; ++t) acc[t] = (v8f){};

  for (int kk = 0; kk < N32; kk += 32) {
    // A 16x32 f16 fragment: lane row = mloc; K runs half*8+0..7 and +16.
    Frag af;
    af.h8[0] = __builtin_nontemporal_load(
        reinterpret_cast<const v8h*>(Arow + kk + half * 8));
    af.h8[1] = __builtin_nontemporal_load(
        reinterpret_cast<const v8h*>(Arow + kk + 16 + half * 8));
    #pragma unroll
    for (int t = 0; t < NT; ++t) {
      // B 32x16 f16 fragment: lane col = mloc; K runs half*8+0..7 and +16
      // in the transposed Z.
      const _Float16* zb = Zcol + (size_t)t * 16 * N32 + kk + half * 8;
      Frag bf;
      bf.h8[0] = *reinterpret_cast<const v8h*>(zb);
      bf.h8[1] = *reinterpret_cast<const v8h*>(zb + 16);
      acc[t] = __builtin_amdgcn_wmma_f32_16x16x32_f16(
          /*neg_a=*/false, af.v, /*neg_b=*/false, bf.v,
          /*c_mod=*/(short)0, acc[t], /*reuse_a=*/false, /*reuse_b=*/false);
    }
  }

  // C/D layout: lane col = mloc, VGPR v -> row M = half*8 + v.
  #pragma unroll
  for (int t = 0; t < NT; ++t) {
    const int col = t * 16 + mloc;
    const float bcol = bias[col];
    #pragma unroll
    for (int v = 0; v < 8; ++v) {
      int r = tm * 16 + half * 8 + v;
      if (r < N) {
        float d = dis[(size_t)b * N + r];
        float hv = d * acc[t][v] + bcol;
        H[((size_t)b * N + r) * F + col] = hv > 0.0f ? hv : 0.0f;   // relu
      }
    }
  }
}

// ---------------- 6. per-feature mean / rstd over (B,N) --------------------
__global__ void bn_stats(const float* __restrict__ H, float* __restrict__ stats,
                         int count, int F) {
  const int f = blockIdx.x;
  const int t = threadIdx.x;
  float s = 0.0f, s2 = 0.0f;
  for (int i = t; i < count; i += blockDim.x) {
    float v = H[(size_t)i * F + f];
    s += v; s2 += v * v;
  }
  __shared__ float ss[256], sq[256];
  ss[t] = s; sq[t] = s2;
  __syncthreads();
  for (int st = 128; st > 0; st >>= 1) {
    if (t < st) { ss[t] += ss[t + st]; sq[t] += sq[t + st]; }
    __syncthreads();
  }
  if (t == 0) {
    float mu  = ss[0] / count;
    float var = sq[0] / count - mu * mu;
    stats[f]       = mu;
    stats[128 + f] = rsqrtf(var + 1e-5f);
  }
}

// ---------------- 7. normalize in place + score projection -----------------
__global__ void norm_score(float* __restrict__ H, const float* __restrict__ stats,
                           const float* __restrict__ g, const float* __restrict__ be,
                           const float* __restrict__ p, float* __restrict__ score,
                           int F, size_t total) {
  size_t t = (size_t)blockIdx.x * blockDim.x + threadIdx.x;
  if (t >= total) return;                   // t = b*N + i
  float* h = H + t * (size_t)F;
  float pn = 0.0f;
  for (int f = 0; f < F; ++f) pn += p[f] * p[f];
  pn = sqrtf(pn);
  float s = 0.0f;
  for (int f = 0; f < F; ++f) {
    float v = (h[f] - stats[f]) * stats[128 + f] * g[f] + be[f];
    h[f] = v;
    s += v * p[f];
  }
  score[t] = s / pn;
}

// ---------------- 8. per-batch top-k (bitonic) + gather X ------------------
__global__ void topk_gather(const float* __restrict__ score, const float* __restrict__ H,
                            int* __restrict__ idxout, float* __restrict__ Xout,
                            int N, int K, int F) {
  const int b = blockIdx.x;
  const int t = threadIdx.x;                // 512 threads
  __shared__ float key[512];
  __shared__ int   ki[512];
  key[t] = (t < N) ? score[(size_t)b * N + t] : NEG_INF;
  ki[t]  = t;
  __syncthreads();
  for (int size = 2; size <= 512; size <<= 1) {
    for (int stride = size >> 1; stride > 0; stride >>= 1) {
      int partner = t ^ stride;
      if (partner > t) {
        bool desc = ((t & size) == 0);
        float a = key[t], c = key[partner];
        if (desc ? (a < c) : (a > c)) {
          key[t] = c; key[partner] = a;
          int tmp = ki[t]; ki[t] = ki[partner]; ki[partner] = tmp;
        }
      }
      __syncthreads();
    }
  }
  if (t < K) idxout[(size_t)b * K + t] = ki[t];
  // Xn[b,r,f] = h[b, idx[r], f] * tanh(val[r])
  for (int e = t; e < K * F; e += 512) {
    int r = e / F, f = e % F;
    float tv = tanhf(key[r]);
    Xout[((size_t)b * K + r) * F + f] = H[((size_t)b * N + ki[r]) * F + f] * tv;
  }
}

// ---------------- 9. A_next[b,r,c] = A[b, idx[r], idx[c]] ------------------
__global__ void gather_A(const float* __restrict__ A, const int* __restrict__ idx,
                         float* __restrict__ An, int N, int K, size_t total) {
  size_t t = (size_t)blockIdx.x * blockDim.x + threadIdx.x;
  if (t >= total) return;
  int c = (int)(t % K);
  size_t r0 = t / K;
  int r = (int)(r0 % K);
  int b = (int)(r0 / K);
  const int* ib = idx + (size_t)b * K;
  An[t] = A[((size_t)b * N + ib[r]) * N + ib[c]];
}

// ---------------- 10. out[b,f] = max_i X[b,i,f] ----------------------------
__global__ void final_max(const float* __restrict__ X, float* __restrict__ out,
                          int K, int F, size_t total) {
  size_t t = (size_t)blockIdx.x * blockDim.x + threadIdx.x;
  if (t >= total) return;
  int f = (int)(t % F);
  int b = (int)(t / F);
  float m = NEG_INF;
  for (int i = 0; i < K; ++i) {
    float v = X[((size_t)b * K + i) * F + f];
    m = v > m ? v : m;
  }
  out[t] = m;
}

// ---------------------------------------------------------------------------
static inline int nblk(size_t n) { return (int)((n + 255) / 256); }

extern "C" void kernel_launch(void* const* d_in, const int* in_sizes, int n_in,
                              void* d_out, int out_size, void* d_ws, size_t ws_size,
                              hipStream_t stream) {
  (void)in_sizes; (void)n_in; (void)out_size; (void)ws_size;

  const float* data = (const float*)d_in[0];
  const float* Wm[3] = {(const float*)d_in[1],  (const float*)d_in[6],  (const float*)d_in[11]};
  const float* bm[3] = {(const float*)d_in[2],  (const float*)d_in[7],  (const float*)d_in[12]};
  const float* gm[3] = {(const float*)d_in[3],  (const float*)d_in[8],  (const float*)d_in[13]};
  const float* em[3] = {(const float*)d_in[4],  (const float*)d_in[9],  (const float*)d_in[14]};
  const float* pm[3] = {(const float*)d_in[5],  (const float*)d_in[10], (const float*)d_in[15]};

  // ---- workspace layout (256B aligned) ----
  char* ws = (char*)d_ws;
  size_t off = 0;
  auto take = [&](size_t bytes) -> void* {
    void* p = ws + off;
    off = (off + bytes + 255) & ~(size_t)255;
    return p;
  };
  float*    A0    = (float*)   take((size_t)BATCH * 400 * 400 * 4);  // 164 MB
  float*    A1    = (float*)   take((size_t)BATCH * 200 * 200 * 4);  //  41 MB
  _Float16* Ahalf = (_Float16*)take((size_t)BATCH * 416 * 416 * 2);  //  89 MB (padded)
  float*    Xa    = (float*)   take((size_t)BATCH * 6400 * 4);       // K*F == 6400 all layers
  float*    Xb    = (float*)   take((size_t)BATCH * 6400 * 4);
  _Float16* Zt    = (_Float16*)take((size_t)BATCH * 16384 * 2);      // max F*N32 = 128*128
  float*    Hbuf  = (float*)   take((size_t)BATCH * 12800 * 4);      // N*F == 12800 all layers
  float*    dis   = (float*)   take((size_t)BATCH * 400 * 4);
  float*    sc    = (float*)   take((size_t)BATCH * 400 * 4);
  int*      idxb  = (int*)     take((size_t)BATCH * 200 * 4);
  float*    stats = (float*)   take(256 * 4);

  const int Ns[3]   = {400, 200, 100};
  const int Fins[3] = {1, 32, 64};
  const int Fos[3]  = {32, 64, 128};
  const int Ks[3]   = {200, 100, 50};

  // Build dense symmetric adjacency (zero diagonal).
  {
    size_t n = (size_t)BATCH * 400 * 400;
    build_adj<<<nblk(n), 256, 0, stream>>>(data, A0, 400, 400 * 399 / 2, n);
  }

  float* Acur = A0;
  float* Anxt = A1;
  const float* Xin = nullptr;   // layer-1 X == ones
  float* Xout = Xa;

  for (int l = 0; l < 3; ++l) {
    const int N = Ns[l], Fin = Fins[l], F = Fos[l], K = Ks[l];
    const int N32 = (N + 31) / 32 * 32;
    {
      size_t n = (size_t)BATCH * N;
      deg_dis<<<nblk(n), 256, 0, stream>>>(Acur, dis, N, n);
    }
    {
      size_t n = (size_t)BATCH * N32 * N32;
      a_to_half_pad<<<nblk(n), 256, 0, stream>>>(Acur, Ahalf, N, N32, n);
    }
    {
      size_t n = (size_t)BATCH * F * N32;
      xw_scale_t<<<nblk(n), 256, 0, stream>>>(Xin, Wm[l], dis, Zt, N, N32, Fin, F, n);
    }
    {
      dim3 grid((N + 15) / 16, BATCH);
      if (F == 32)
        gcn_agg_wmma<2><<<grid, 32, 0, stream>>>(Ahalf, Zt, dis, bm[l], Hbuf, N, N32);
      else if (F == 64)
        gcn_agg_wmma<4><<<grid, 32, 0, stream>>>(Ahalf, Zt, dis, bm[l], Hbuf, N, N32);
      else
        gcn_agg_wmma<8><<<grid, 32, 0, stream>>>(Ahalf, Zt, dis, bm[l], Hbuf, N, N32);
    }
    bn_stats<<<F, 256, 0, stream>>>(Hbuf, stats, BATCH * N, F);
    {
      size_t n = (size_t)BATCH * N;
      norm_score<<<nblk(n), 256, 0, stream>>>(Hbuf, stats, gm[l], em[l], pm[l], sc, F, n);
    }
    topk_gather<<<BATCH, 512, 0, stream>>>(sc, Hbuf, idxb, Xout, N, K, F);
    {
      size_t n = (size_t)BATCH * K * K;
      gather_A<<<nblk(n), 256, 0, stream>>>(Acur, idxb, Anxt, N, K, n);
    }
    float* tA = Acur; Acur = Anxt; Anxt = tA;
    Xin = Xout; Xout = (Xout == Xa) ? Xb : Xa;
  }

  {
    size_t n = (size_t)BATCH * 128;
    final_max<<<nblk(n), 256, 0, stream>>>(Xin, (float*)d_out, 50, 128, n);
  }
}
